// transformer_71554155151496
// MI455X (gfx1250) — matrix operand order repaired and checked by reading, not compiled, not run on hardware
//
#include <hip/hip_runtime.h>
#include <hip/hip_bf16.h>
#include <math.h>

// ---------------------------------------------------------------------------
// Transformer enc-dec forward for MI455X (gfx1250, wave32, WMMA f16->f32).
// All matmuls go through v_wmma_f32_16x16x32_f16.  A-tile staging uses the
// CDNA5 async global->LDS path when the toolchain declares the builtins.
// ---------------------------------------------------------------------------

typedef _Float16 v8h  __attribute__((ext_vector_type(8)));
typedef _Float16 v16h __attribute__((ext_vector_type(16)));
typedef float    v8f  __attribute__((ext_vector_type(8)));
typedef int      v4i_gcc __attribute__((vector_size(16)));  // matches builtin

#define TD   2048          // tokens = B*L
#define DD   512           // hidden
#define FFD  2048          // ffn hidden
#define VOC  16000
#define LSEQ 256
#define NH   8
#define DK   64

#if __has_builtin(__builtin_amdgcn_global_load_async_to_lds_b128) && \
    __has_builtin(__builtin_amdgcn_s_wait_asynccnt)
#define USE_ASYNC_LDS 1
#else
#define USE_ASYNC_LDS 0
#endif

#if USE_ASYNC_LDS
__device__ __forceinline__ void async_copy_b128(const void* g, void* l) {
  __builtin_amdgcn_global_load_async_to_lds_b128(
      (__attribute__((address_space(1))) v4i_gcc*)g,
      (__attribute__((address_space(3))) v4i_gcc*)l, 0, 0);
}
#endif

__device__ __forceinline__ v8f wmma_f16(v16h a, v16h b, v8f c) {
  return __builtin_amdgcn_wmma_f32_16x16x32_f16(false, a, false, b, (short)0, c,
                                                false, false);
}

// A fragment: 16x32 tile at `base` (row-major, leading dim ld halves), K window
// starting at kb.  lane<16: row=lane, K = kb+{0..7, 16..23};
// lane>=16: row=lane-16, K = kb+{8..15, 24..31}.  Two contiguous 16B loads.
__device__ __forceinline__ v16h a_frag(const _Float16* base, int ld, int kb,
                                       int lane) {
  int r = lane & 15, hs = (lane >> 4) * 8;
  const _Float16* p = base + (size_t)r * ld + kb + hs;
  v8h lo = *(const v8h*)p;
  v8h hi = *(const v8h*)(p + 16);
  return __builtin_shufflevector(lo, hi, 0, 1, 2, 3, 4, 5, 6, 7, 8, 9, 10, 11,
                                 12, 13, 14, 15);
}

// B fragment: B[k][n] sourced from row-major storage S[n][k] (n = 16 rows,
// leading dim ld).  lane<16: n=lane, K=kb+{0..7, 16..23}; lane>=16: n=lane-16,
// K=kb+{8..15, 24..31}.  Two contiguous 16B loads.
__device__ __forceinline__ v16h b_frag(const _Float16* S, int ld, int kb,
                                       int lane) {
  int n = lane & 15, ko = (lane >> 4) * 8;
  const _Float16* p = S + (size_t)n * ld + kb + ko;
  v8h lo = *(const v8h*)p;
  v8h hi = *(const v8h*)(p + 16);
  return __builtin_shufflevector(lo, hi, 0, 1, 2, 3, 4, 5, 6, 7, 8, 9, 10, 11,
                                 12, 13, 14, 15);
}

// ---------------------------------------------------------------------------
// GEMM:  C[M,N] = act( A_f16[M,K] @ W_f32[N,K]^T + bias[N] )
// Block tile 128x128, 8 waves (4 along M x 2 along N), each wave 32x64
// (2x4 WMMA accumulators), BK=32.  8 WMMAs per 6 fragment loads per k-step.
// ---------------------------------------------------------------------------
__global__ __launch_bounds__(256) void gemm_kernel(
    const _Float16* __restrict__ A, const float* __restrict__ W,
    const float* __restrict__ bias, float* __restrict__ Cf,
    _Float16* __restrict__ Ch, int M, int N, int K, int relu) {
  __shared__ alignas(16) _Float16 As[128 * 32];   // 8KB
  __shared__ alignas(16) _Float16 Ws[128 * 32];   // 8KB
  int bm = blockIdx.x * 128, bn = blockIdx.y * 128;
  int tid = threadIdx.x, wave = tid >> 5, lane = tid & 31;
  int wm = (wave & 3) * 32, wn = (wave >> 2) * 64;

  v8f zero = {0.f, 0.f, 0.f, 0.f, 0.f, 0.f, 0.f, 0.f};
  v8f acc[2][4];
#pragma unroll
  for (int i = 0; i < 2; i++)
#pragma unroll
    for (int j = 0; j < 4; j++) acc[i][j] = zero;

  int ar = tid >> 1, ac = (tid & 1) * 16;   // A stage: 16 halves / thread
  int wr = tid >> 1, wc = (tid & 1) * 16;   // W stage: 16 floats / thread

  for (int k0 = 0; k0 < K; k0 += 32) {
    // stage A tile 128x32 f16 (contiguous -> async global->LDS if available)
    {
      const _Float16* src = A + (size_t)(bm + ar) * K + k0 + ac;
#if USE_ASYNC_LDS
      async_copy_b128(src, &As[ar * 32 + ac]);
      async_copy_b128(src + 8, &As[ar * 32 + ac + 8]);
#else
      *(v8h*)&As[ar * 32 + ac]     = *(const v8h*)src;
      *(v8h*)&As[ar * 32 + ac + 8] = *(const v8h*)(src + 8);
#endif
    }
    // stage W tile 128x32, f32 -> f16 convert in flight (16 floats / thread)
    {
      const float* src = W + (size_t)(bn + wr) * K + k0 + wc;
      v8h t0, t1;
#pragma unroll
      for (int e = 0; e < 8; e++) {
        t0[e] = (_Float16)src[e];
        t1[e] = (_Float16)src[8 + e];
      }
      *(v8h*)&Ws[wr * 32 + wc]     = t0;
      *(v8h*)&Ws[wr * 32 + wc + 8] = t1;
    }
#if USE_ASYNC_LDS
    __builtin_amdgcn_s_wait_asynccnt(0);
#endif
    __syncthreads();

    v16h bf[4];
#pragma unroll
    for (int j = 0; j < 4; j++)
      bf[j] = b_frag(&Ws[(wn + j * 16) * 32], 32, 0, lane);
#pragma unroll
    for (int i = 0; i < 2; i++) {
      v16h a = a_frag(&As[(wm + i * 16) * 32], 32, 0, lane);
#pragma unroll
      for (int j = 0; j < 4; j++) acc[i][j] = wmma_f16(a, bf[j], acc[i][j]);
    }
    __syncthreads();
  }

  int nl = lane & 15, hi = lane >> 4;
#pragma unroll
  for (int i = 0; i < 2; i++)
#pragma unroll
    for (int j = 0; j < 4; j++) {
      int col = bn + wn + j * 16 + nl;
      float bv = bias ? bias[col] : 0.f;
#pragma unroll
      for (int r = 0; r < 8; r++) {
        int row = bm + wm + i * 16 + hi * 8 + r;
        float v = acc[i][j][r] + bv;
        if (relu) v = fmaxf(v, 0.f);
        size_t o = (size_t)row * N + col;
        if (Cf) Cf[o] = v;
        if (Ch) Ch[o] = (_Float16)v;
      }
    }
}

// ---------------------------------------------------------------------------
// Attention part 1: scores + mask + softmax -> P (f16, [B*H, 256, 256]).
// Head h lives in columns d*8+h of the projection (interleaved head split).
// grid (4 qchunks, H, B), 128 threads (wave w handles q-tile qc*64+w*16).
// ---------------------------------------------------------------------------
__global__ __launch_bounds__(128) void attn_score_kernel(
    const _Float16* __restrict__ Qf, const _Float16* __restrict__ Kf,
    const int* __restrict__ ktok, _Float16* __restrict__ P, int causal,
    float scale) {
  __shared__ alignas(16) _Float16 Kh[LSEQ * DK];  // 32KB, Kh[key][d]
  int qc = blockIdx.x, h = blockIdx.y, b = blockIdx.z;
  int tid = threadIdx.x, wave = tid >> 5, lane = tid & 31;

  for (int idx = tid; idx < LSEQ * DK; idx += 128) {
    int l = idx >> 6, d = idx & 63;
    Kh[idx] = Kf[((size_t)(b * LSEQ + l)) * DD + d * NH + h];
  }
  __syncthreads();

  int q0 = qc * 64 + wave * 16;
  int nl = lane & 15, hi = lane >> 4;

  // Q A-fragments gathered from global (stride-8 head interleave)
  v16h aq[2];
#pragma unroll
  for (int s = 0; s < 2; s++) {
    const _Float16* qrow = Qf + ((size_t)(b * LSEQ + q0 + nl)) * DD + h;
    v16h a;
#pragma unroll
    for (int e = 0; e < 8; e++) {
      int d0 = s * 32 + hi * 8 + e;
      a[e]     = qrow[(size_t)d0 * NH];
      a[e + 8] = qrow[(size_t)(d0 + 16) * NH];
    }
    aq[s] = a;
  }

  v8f S[16];
  v8f zero = {0.f, 0.f, 0.f, 0.f, 0.f, 0.f, 0.f, 0.f};
#pragma unroll
  for (int t = 0; t < 16; t++) {
    v16h b0 = b_frag(&Kh[(t * 16) * DK], DK, 0, lane);
    v16h b1 = b_frag(&Kh[(t * 16) * DK], DK, 32, lane);
    v8f acc = zero;
    acc = wmma_f16(aq[0], b0, acc);
    acc = wmma_f16(aq[1], b1, acc);
    S[t] = acc;
  }

  // mask + scale, track row maxima.  Element S[t][r]: key=t*16+nl,
  // query row = q0 + r + 8*hi.
  float m[8];
#pragma unroll
  for (int r = 0; r < 8; r++) m[r] = -3.0e38f;
#pragma unroll
  for (int t = 0; t < 16; t++) {
    int key = t * 16 + nl;
    int kv = ktok[b * LSEQ + key];
#pragma unroll
    for (int r = 0; r < 8; r++) {
      int qrow = q0 + r + 8 * hi;
      float v = S[t][r] * scale;
      if (kv == 0 || (causal && key > qrow)) v = -1e10f;
      S[t][r] = v;
      m[r] = fmaxf(m[r], v);
    }
  }
#pragma unroll
  for (int r = 0; r < 8; r++) {
    m[r] = fmaxf(m[r], __shfl_xor(m[r], 1, 32));
    m[r] = fmaxf(m[r], __shfl_xor(m[r], 2, 32));
    m[r] = fmaxf(m[r], __shfl_xor(m[r], 4, 32));
    m[r] = fmaxf(m[r], __shfl_xor(m[r], 8, 32));
  }
  float sum[8];
#pragma unroll
  for (int r = 0; r < 8; r++) sum[r] = 0.f;
#pragma unroll
  for (int t = 0; t < 16; t++)
#pragma unroll
    for (int r = 0; r < 8; r++) {
      float p = __expf(S[t][r] - m[r]);
      S[t][r] = p;
      sum[r] += p;
    }
#pragma unroll
  for (int r = 0; r < 8; r++) {
    sum[r] += __shfl_xor(sum[r], 1, 32);
    sum[r] += __shfl_xor(sum[r], 2, 32);
    sum[r] += __shfl_xor(sum[r], 4, 32);
    sum[r] += __shfl_xor(sum[r], 8, 32);
    sum[r] = 1.f / sum[r];
  }

  _Float16* Pb = P + ((size_t)((b * NH + h) * LSEQ)) * LSEQ;
#pragma unroll
  for (int t = 0; t < 16; t++)
#pragma unroll
    for (int r = 0; r < 8; r++)
      Pb[(size_t)(q0 + r + 8 * hi) * LSEQ + t * 16 + nl] =
          (_Float16)(S[t][r] * sum[r]);
}

// ---------------------------------------------------------------------------
// Attention part 2: O = P @ Vhead, written head-major concatenated
// (column h*64+d of Acat).  V staged transposed in LDS for contiguous B-frags.
// ---------------------------------------------------------------------------
__global__ __launch_bounds__(128) void attn_pv_kernel(
    const _Float16* __restrict__ P, const _Float16* __restrict__ Vf,
    _Float16* __restrict__ Acat) {
  __shared__ alignas(16) _Float16 Vt[DK * LSEQ];  // 32KB, Vt[d][key]
  int qc = blockIdx.x, h = blockIdx.y, b = blockIdx.z;
  int tid = threadIdx.x, wave = tid >> 5, lane = tid & 31;

  for (int idx = tid; idx < DK * LSEQ; idx += 128) {
    int l = idx & 255, d = idx >> 8;
    Vt[d * LSEQ + l] = Vf[((size_t)(b * LSEQ + l)) * DD + d * NH + h];
  }
  __syncthreads();

  int q0 = qc * 64 + wave * 16;
  const _Float16* Pb = P + ((size_t)((b * NH + h) * LSEQ + q0)) * LSEQ;

  v8f zero = {0.f, 0.f, 0.f, 0.f, 0.f, 0.f, 0.f, 0.f};
  v8f O[4];
  O[0] = zero; O[1] = zero; O[2] = zero; O[3] = zero;

#pragma unroll
  for (int s = 0; s < 8; s++) {
    v16h ap = a_frag(Pb, LSEQ, s * 32, lane);
#pragma unroll
    for (int j = 0; j < 4; j++) {
      v16h bv = b_frag(&Vt[(j * 16) * LSEQ], LSEQ, s * 32, lane);
      O[j] = wmma_f16(ap, bv, O[j]);
    }
  }

  int nl = lane & 15, hi = lane >> 4;
#pragma unroll
  for (int j = 0; j < 4; j++)
#pragma unroll
    for (int r = 0; r < 8; r++)
      Acat[((size_t)(b * LSEQ + q0 + r + 8 * hi)) * DD + h * DK + j * 16 + nl] =
          (_Float16)O[j][r];
}

// ---------------------------------------------------------------------------
// Embedding:  s = npm * (2*U[tok] + pos[l]);  writes f32 + f16 mirror.
// ---------------------------------------------------------------------------
__global__ __launch_bounds__(256) void embed_kernel(
    const int* __restrict__ tok, const float* __restrict__ U,
    const float* __restrict__ pos, float* __restrict__ S,
    _Float16* __restrict__ Sh) {
  int row = blockIdx.x, tid = threadIdx.x;
  int t = tok[row];
  int l = row & (LSEQ - 1);
  float npm = (t != 0) ? 1.f : 0.f;
  for (int c = tid; c < DD; c += 256) {
    float v = npm * (2.f * U[(size_t)t * DD + c] + pos[(size_t)l * DD + c]);
    S[(size_t)row * DD + c] = v;
    Sh[(size_t)row * DD + c] = (_Float16)v;
  }
}

// ---------------------------------------------------------------------------
// LayerNorm(Y+R) with ddof=1 and eps=1e-8, fused *npm; writes f32 + f16.
// One 256-thread block per token row.
// ---------------------------------------------------------------------------
__global__ __launch_bounds__(256) void add_norm_kernel(
    const float* __restrict__ Y, const float* __restrict__ R,
    const int* __restrict__ tok, const float* __restrict__ gp,
    const float* __restrict__ bp, float* __restrict__ Of,
    _Float16* __restrict__ Oh) {
  __shared__ float red[256];
  int row = blockIdx.x, tid = threadIdx.x;
  size_t base = (size_t)row * DD;
  float a0 = Y[base + tid] + R[base + tid];
  float a1 = Y[base + tid + 256] + R[base + tid + 256];
  red[tid] = a0 + a1;
  __syncthreads();
  for (int o = 128; o > 0; o >>= 1) {
    if (tid < o) red[tid] += red[tid + o];
    __syncthreads();
  }
  float mean = red[0] * (1.f / 512.f);
  __syncthreads();
  float d0 = a0 - mean, d1 = a1 - mean;
  red[tid] = d0 * d0 + d1 * d1;
  __syncthreads();
  for (int o = 128; o > 0; o >>= 1) {
    if (tid < o) red[tid] += red[tid + o];
    __syncthreads();
  }
  float var = red[0] * (1.f / 511.f);  // unbiased (ddof=1)
  float rs = rsqrtf(var + 1e-8f);
  float g = gp[0], bb = bp[0];
  float npm = (tok[row] != 0) ? 1.f : 0.f;
  float o0 = (g * d0 * rs + bb) * npm;
  float o1 = (g * d1 * rs + bb) * npm;
  Of[base + tid] = o0;
  Of[base + tid + 256] = o1;
  if (Oh) {
    Oh[base + tid] = (_Float16)o0;
    Oh[base + tid + 256] = (_Float16)o1;
  }
}

// s += xn; refresh f16 mirror of s.
__global__ __launch_bounds__(256) void accum_kernel(
    float* __restrict__ s, const float* __restrict__ xn,
    _Float16* __restrict__ sh) {
  size_t i = (size_t)blockIdx.x * 256 + threadIdx.x;
  float v = s[i] + xn[i];
  s[i] = v;
  sh[i] = (_Float16)v;
}

// Row softmax over VOC columns, in place on d_out.
__global__ __launch_bounds__(256) void softmax_kernel(float* __restrict__ O,
                                                      int ncol) {
  __shared__ float red[256];
  int row = blockIdx.x, tid = threadIdx.x;
  float* p = O + (size_t)row * ncol;
  float m = -3.0e38f;
  for (int c = tid; c < ncol; c += 256) m = fmaxf(m, p[c]);
  red[tid] = m;
  __syncthreads();
  for (int o = 128; o > 0; o >>= 1) {
    if (tid < o) red[tid] = fmaxf(red[tid], red[tid + o]);
    __syncthreads();
  }
  m = red[0];
  __syncthreads();
  float s = 0.f;
  for (int c = tid; c < ncol; c += 256) s += __expf(p[c] - m);
  red[tid] = s;
  __syncthreads();
  for (int o = 128; o > 0; o >>= 1) {
    if (tid < o) red[tid] += red[tid + o];
    __syncthreads();
  }
  float inv = 1.f / red[0];
  __syncthreads();
  for (int c = tid; c < ncol; c += 256) p[c] = __expf(p[c] - m) * inv;
}

// ---------------------------------------------------------------------------
// Host orchestration.
// Leaf order assumption: top-level dict in insertion order
//   (x, y, params, pos_x, pos_y, causal); params flattened jax-tree style
//   (dict keys alphabetical, lists in order):
//   Ud, Ue, Vd, dec[6]{ca(10), ffn(6), sa(10)}, enc[6]{ffn(6), sa(10)}
//   mha leaves: b,bk,bo,bq,bv,g,wk,wo,wq,wv ; ffn leaves: b,b1,b2,g,w1,w2
// ---------------------------------------------------------------------------
extern "C" void kernel_launch(void* const* d_in, const int* in_sizes, int n_in,
                              void* d_out, int out_size, void* d_ws,
                              size_t ws_size, hipStream_t stream) {
  const int* xt = (const int*)d_in[0];
  const int* yt = (const int*)d_in[1];
  auto F = [&](int i) { return (const float*)d_in[i]; };

  struct Mha { const float *wq, *bq, *wk, *bk, *wv, *bv, *wo, *bo, *g, *b; };
  struct Ffn { const float *w1, *b1, *w2, *b2, *g, *b; };
  auto mha_at = [&](int base) {
    Mha m;
    m.b = F(base + 0); m.bk = F(base + 1); m.bo = F(base + 2);
    m.bq = F(base + 3); m.bv = F(base + 4); m.g = F(base + 5);
    m.wk = F(base + 6); m.wo = F(base + 7); m.wq = F(base + 8);
    m.wv = F(base + 9);
    return m;
  };
  auto ffn_at = [&](int base) {
    Ffn f;
    f.b = F(base + 0); f.b1 = F(base + 1); f.b2 = F(base + 2);
    f.g = F(base + 3); f.w1 = F(base + 4); f.w2 = F(base + 5);
    return f;
  };

  const float* Ud = F(2);
  const float* Ue = F(3);
  const float* Vd = F(4);
  // dec layer i: base 5 + i*26  (ca +0, ffn +10, sa +16)
  // enc layer i: base 161 + i*16 (ffn +0, sa +6)
  const float* pos_x = F(257);
  const float* pos_y = F(258);

  // --- workspace bump allocator ---
  char* wp = (char*)d_ws;
  size_t off = 0;
  auto alloc = [&](size_t bytes) -> void* {
    off = (off + 255) & ~(size_t)255;
    void* p = wp + off;
    off += bytes;
    return p;
  };
  const size_t TDb = (size_t)TD * DD;
  float* s  = (float*)alloc(TDb * 4);
  float* t1 = (float*)alloc(TDb * 4);
  float* t2 = (float*)alloc(TDb * 4);
  float* t3 = (float*)alloc(TDb * 4);
  float* t4 = (float*)alloc(TDb * 4);
  _Float16* sf16   = (_Float16*)alloc(TDb * 2);
  _Float16* t2f16  = (_Float16*)alloc(TDb * 2);
  _Float16* t4f16  = (_Float16*)alloc(TDb * 2);
  _Float16* acat   = (_Float16*)alloc(TDb * 2);
  _Float16* qf16   = (_Float16*)alloc(TDb * 2);
  _Float16* kf16   = (_Float16*)alloc(TDb * 2);
  _Float16* vf16   = (_Float16*)alloc(TDb * 2);
  _Float16* enf16  = (_Float16*)alloc(TDb * 2);
  _Float16* ynf16  = (_Float16*)alloc(TDb * 2);
  _Float16* hf16   = (_Float16*)alloc((size_t)TD * FFD * 2);
  _Float16* Pbuf   = (_Float16*)alloc((size_t)NH * 8 * LSEQ * LSEQ * 2);

  const float scale = (float)(1.0 / sqrt(512.0));

  auto gemm = [&](const _Float16* A, const float* W, const float* bias,
                  float* Cf, _Float16* Ch, int M, int N, int K, int relu) {
    dim3 g((unsigned)(M / 128), (unsigned)(N / 128));
    gemm_kernel<<<g, 256, 0, stream>>>(A, W, bias, Cf, Ch, M, N, K, relu);
  };
  dim3 agrid(4, NH, 8);

  // ================= encoder =================
  embed_kernel<<<TD, 256, 0, stream>>>(xt, Ue, pos_x, s, sf16);
  for (int i = 0; i < 6; i++) {
    Mha sa = mha_at(161 + i * 16 + 6);
    Ffn ff = ffn_at(161 + i * 16 + 0);
    gemm(sf16, sa.wq, sa.bq, nullptr, qf16, TD, DD, DD, 0);
    gemm(sf16, sa.wk, sa.bk, nullptr, kf16, TD, DD, DD, 0);
    gemm(sf16, sa.wv, sa.bv, nullptr, vf16, TD, DD, DD, 0);
    attn_score_kernel<<<agrid, 128, 0, stream>>>(qf16, kf16, xt, Pbuf, 0, scale);
    attn_pv_kernel<<<agrid, 128, 0, stream>>>(Pbuf, vf16, acat);
    gemm(acat, sa.wo, sa.bo, t1, nullptr, TD, DD, DD, 0);
    add_norm_kernel<<<TD, 256, 0, stream>>>(t1, s, xt, sa.g, sa.b, t2, t2f16);
    gemm(t2f16, ff.w1, ff.b1, nullptr, hf16, TD, FFD, DD, 1);
    gemm(hf16, ff.w2, ff.b2, t1, nullptr, TD, DD, FFD, 0);
    add_norm_kernel<<<TD, 256, 0, stream>>>(t1, t2, xt, ff.g, ff.b, t3, enf16);
    accum_kernel<<<TD * DD / 256, 256, 0, stream>>>(s, t3, sf16);
  }
  // enf16 now holds f16(en) = last encoder layer output (already *npm_x)

  // ================= decoder =================
  embed_kernel<<<TD, 256, 0, stream>>>(yt, Ud, pos_y, s, sf16);
  for (int i = 0; i < 6; i++) {
    Mha sa = mha_at(5 + i * 26 + 16);
    Mha ca = mha_at(5 + i * 26 + 0);
    Ffn ff = ffn_at(5 + i * 26 + 10);
    // masked self-attention (pad(y) | causal)
    gemm(sf16, sa.wq, sa.bq, nullptr, qf16, TD, DD, DD, 0);
    gemm(sf16, sa.wk, sa.bk, nullptr, kf16, TD, DD, DD, 0);
    gemm(sf16, sa.wv, sa.bv, nullptr, vf16, TD, DD, DD, 0);
    attn_score_kernel<<<agrid, 128, 0, stream>>>(qf16, kf16, yt, Pbuf, 1, scale);
    attn_pv_kernel<<<agrid, 128, 0, stream>>>(Pbuf, vf16, acat);
    gemm(acat, sa.wo, sa.bo, t1, nullptr, TD, DD, DD, 0);
    add_norm_kernel<<<TD, 256, 0, stream>>>(t1, s, yt, sa.g, sa.b, t2, t2f16);
    // cross-attention (queries from out1, keys/values from encoder)
    gemm(t2f16, ca.wq, ca.bq, nullptr, qf16, TD, DD, DD, 0);
    gemm(enf16, ca.wk, ca.bk, nullptr, kf16, TD, DD, DD, 0);
    gemm(enf16, ca.wv, ca.bv, nullptr, vf16, TD, DD, DD, 0);
    attn_score_kernel<<<agrid, 128, 0, stream>>>(qf16, kf16, xt, Pbuf, 0, scale);
    attn_pv_kernel<<<agrid, 128, 0, stream>>>(Pbuf, vf16, acat);
    gemm(acat, ca.wo, ca.bo, t1, nullptr, TD, DD, DD, 0);
    add_norm_kernel<<<TD, 256, 0, stream>>>(t1, t2, yt, ca.g, ca.b, t4, t4f16);
    // ffn
    gemm(t4f16, ff.w1, ff.b1, nullptr, hf16, TD, FFD, DD, 1);
    gemm(hf16, ff.w2, ff.b2, t1, nullptr, TD, DD, FFD, 0);
    add_norm_kernel<<<TD, 256, 0, stream>>>(t1, t4, yt, ff.g, ff.b, t3, ynf16);
    accum_kernel<<<TD * DD / 256, 256, 0, stream>>>(s, t3, sf16);
  }

  // logits = yn @ Vd^T  (no bias), then row softmax over vocab, in d_out.
  float* out = (float*)d_out;
  gemm(ynf16, Vd, nullptr, out, nullptr, TD, VOC, DD, 0);
  softmax_kernel<<<TD, 256, 0, stream>>>(out, VOC);
}
